// Model_9208409882715
// MI455X (gfx1250) — compile-verified, matrix operand order checked by reading, not run on hardware
//
#include <hip/hip_runtime.h>
#include <math.h>

#define VOCAB   65024
#define DMODEL  2048
#define NLAYER  6
#define NH      16
#define HDIM    128
#define FFDIM   8192
#define SEQ     512
#define BATCH   2
#define NTOK    (BATCH * SEQ)   // 1024 token rows

typedef __attribute__((ext_vector_type(16))) __bf16 v16bf;
typedef __attribute__((ext_vector_type(8)))  __bf16 v8bf;
typedef __attribute__((ext_vector_type(4)))  __bf16 v4bf;
typedef __attribute__((ext_vector_type(8)))  float  v8f;

// ---------------------------------------------------------------------------
// WMMA 16x16x32 bf16 operand layout (wave32):
//   lane L<16 : row/col = L, K elements {0..7, 16..23}
//   lane L+16 : row/col = L, K elements {8..15, 24..31}
// Every operand tile lives in LDS in "fragment-ready" form:
//   v16bf Frag[fragId][lane]  ->  one aligned 32B read (2x ds_load_b128).
// frag_pos maps a K index (0..31) to (half-sel h, element index i).
// ---------------------------------------------------------------------------
static __device__ __forceinline__ void frag_pos(int k, int& h, int& i) {
  int kk = k & 15, grp = (k >> 4) & 1;
  h = kk >> 3;
  i = (kk & 7) + grp * 8;
}

static __device__ __forceinline__ float gelu_tanh(float x) {
  const float k = 0.7978845608028654f;   // sqrt(2/pi)
  return 0.5f * x * (1.f + tanhf(k * (x + 0.044715f * x * x * x)));
}

// ---------------------------------------------------------------------------
// Embedding gather: h[row,:] = emb[ids[row],:]
// ---------------------------------------------------------------------------
__global__ __launch_bounds__(256)
void embed_kernel(const int* __restrict__ ids, const float* __restrict__ emb,
                  float* __restrict__ h)
{
  const int row = blockIdx.x;
  const int id  = ids[row];
  const float4* src = (const float4*)(emb + (size_t)id * DMODEL);
  float4*       dst = (float4*)(h + (size_t)row * DMODEL);
  for (int i = threadIdx.x; i < DMODEL / 4; i += 256) dst[i] = src[i];
}

// ---------------------------------------------------------------------------
// LayerNorm, one block per row of length DMODEL
// ---------------------------------------------------------------------------
__global__ __launch_bounds__(256)
void layernorm_kernel(const float* __restrict__ x, const float* __restrict__ g,
                      const float* __restrict__ bta, float* __restrict__ y)
{
  __shared__ float red[256];
  const int row = blockIdx.x;
  const int t = threadIdx.x;
  const float* xr = x + (size_t)row * DMODEL;
  float s = 0.f, s2 = 0.f;
#pragma unroll
  for (int i = 0; i < DMODEL / 256; ++i) {
    float v = xr[t + i * 256];
    s += v; s2 += v * v;
  }
  red[t] = s; __syncthreads();
  for (int o = 128; o > 0; o >>= 1) { if (t < o) red[t] += red[t + o]; __syncthreads(); }
  const float mean = red[0] / DMODEL;
  __syncthreads();
  red[t] = s2; __syncthreads();
  for (int o = 128; o > 0; o >>= 1) { if (t < o) red[t] += red[t + o]; __syncthreads(); }
  const float var = red[0] / DMODEL - mean * mean;
  const float inv = rsqrtf(var + 1e-5f);
  float* yr = y + (size_t)row * DMODEL;
#pragma unroll
  for (int i = 0; i < DMODEL / 256; ++i) {
    int k = t + i * 256;
    yr[k] = (xr[k] - mean) * inv * g[k] + bta[k];
  }
}

// ---------------------------------------------------------------------------
// Y[M,N] = act(X[M,K] @ dequant(Wp,scale)[N,K]^T + bias) + residual
// int4 nibbles: lo nibble -> even k, hi nibble -> odd k, value (nib-8)*scale.
// Block tile 128x64, BK=32, 8 waves; fragment-ready LDS, double-buffered:
// global loads of tile s+1 are issued before the WMMAs of tile s so HBM
// latency hides behind the matrix pipes; one barrier per K-step. All LDS
// fragments are pulled into registers before the WMMA sequence so ds_load
// latency is covered by partial s_wait_dscnt counts, not hard zero-waits.
// ---------------------------------------------------------------------------
__global__ __launch_bounds__(256)
void gemm_dq_wmma(const float* __restrict__ X, const int* __restrict__ Wp,
                  const float* __restrict__ scale, const float* __restrict__ bias,
                  const float* __restrict__ residual, float* __restrict__ Y,
                  int M, int N, int K, int doGelu)
{
  __shared__ v16bf Af[2][8][32];   // [buf][row-tile][lane]  16 KB
  __shared__ v16bf Bf[2][4][32];   // [buf][col-tile][lane]   8 KB

  const int tid  = threadIdx.x;
  const int lane = tid & 31;
  const int wave = tid >> 5;
  const int half = lane >> 4;
  const int lrow = lane & 15;
  const int mBase = blockIdx.y * 128;
  const int nBase = blockIdx.x * 64;
  const int Kb = K >> 1;                    // packed elements per weight row

  const int wn  = tid >> 2;                 // 0..63 : staged output column
  const int wko = (tid & 3) << 2;           // 0,4,8,12 : packed offset in 16
  const float wsc = scale[nBase + wn];
  const int* wrow = Wp + (size_t)(nBase + wn) * Kb;
  int wh, wi0; frag_pos(wko * 2, wh, wi0);  // 8 consecutive k -> one v8bf run

  float4 xr[4];                             // in-flight X tile
  int4   wv;                                // in-flight packed weights

  auto gload = [&](int kb) {
#pragma unroll
    for (int it = 0; it < 4; ++it) {
      int f4 = tid + it * 256;              // 0..1023 float4 slots
      int r  = f4 >> 3;                     // 0..127
      int kq = (f4 & 7) << 2;               // 0,4,...,28
      xr[it] = *(const float4*)(X + (size_t)(mBase + r) * K + kb + kq);
    }
    wv = *(const int4*)(wrow + (kb >> 1) + wko);
    if (kb + 32 < K)                        // global_prefetch_b8 on next tile
      __builtin_prefetch(wrow + ((kb + 32) >> 1) + wko, 0, 1);
  };
  auto stage = [&](int buf) {
#pragma unroll
    for (int it = 0; it < 4; ++it) {
      int f4 = tid + it * 256;
      int r  = f4 >> 3;
      int kq = (f4 & 7) << 2;
      int h, i0; frag_pos(kq, h, i0);
      v4bf t4;
      t4[0] = (__bf16)xr[it].x; t4[1] = (__bf16)xr[it].y;
      t4[2] = (__bf16)xr[it].z; t4[3] = (__bf16)xr[it].w;
      *(v4bf*)((__bf16*)&Af[buf][r >> 4][h * 16 + (r & 15)] + i0) = t4;
    }
    const int pb[4] = {wv.x, wv.y, wv.z, wv.w};
    v8bf t8;
#pragma unroll
    for (int j = 0; j < 4; ++j) {
      int byte = pb[j] & 0xFF;
      t8[2 * j + 0] = (__bf16)((float)((byte & 15) - 8) * wsc);
      t8[2 * j + 1] = (__bf16)((float)(((byte >> 4) & 15) - 8) * wsc);
    }
    *(v8bf*)((__bf16*)&Bf[buf][wn >> 4][wh * 16 + (wn & 15)] + wi0) = t8;
  };

  v8f acc[4] = {};

  // prologue: tile 0 into buffer 0
  gload(0);
  stage(0);

  const int nsteps = K >> 5;
  for (int s = 0; s < nsteps; ++s) {
    __syncthreads();
    const int cur = s & 1;
    const bool more = (s + 1 < nsteps);
    if (more) gload((s + 1) << 5);          // loads in flight during WMMAs

    // pull all fragments first (10x ds_load_b128), then issue 4 WMMAs
    const v16bf a = Af[cur][wave][lane];
    v16bf b[4];
#pragma unroll
    for (int nt = 0; nt < 4; ++nt) b[nt] = Bf[cur][nt][lane];
#pragma unroll
    for (int nt = 0; nt < 4; ++nt)
      acc[nt] = __builtin_amdgcn_wmma_f32_16x16x32_bf16(
          false, a, false, b[nt], (short)0, acc[nt], false, false);

    if (more) stage(cur ^ 1);               // convert + store next tile
  }

  // --- epilogue: bias -> gelu -> residual (matches reference usage) ---
#pragma unroll
  for (int nt = 0; nt < 4; ++nt) {
#pragma unroll
    for (int r = 0; r < 8; ++r) {
      int m = mBase + wave * 16 + half * 8 + r;
      int n = nBase + nt * 16 + lrow;
      float v = acc[nt][r];
      if (bias)     v += bias[n];
      if (doGelu)   v  = gelu_tanh(v);
      if (residual) v += residual[(size_t)m * N + n];
      Y[(size_t)m * N + n] = v;
    }
  }
}

// ---------------------------------------------------------------------------
// RoPE (in place on q and k inside qkv buffer). half = HDIM/2 = 64.
// ---------------------------------------------------------------------------
__global__ __launch_bounds__(64)
void rope_kernel(float* __restrict__ qkv, const int* __restrict__ pos)
{
  const int d  = threadIdx.x;                 // 0..63
  const int hh = blockIdx.x, s = blockIdx.y, b = blockIdx.z;
  const float p   = (float)pos[b * SEQ + s];
  const float inv = powf(10000.f, -(float)d / 64.f);
  const float ang = p * inv;
  const float cs = cosf(ang), sn = sinf(ang);
#pragma unroll
  for (int c = 0; c < 2; ++c) {               // q then k
    size_t base = (((size_t)(b * SEQ + s)) * 3 + c) * DMODEL + hh * HDIM;
    float x1 = qkv[base + d], x2 = qkv[base + 64 + d];
    qkv[base + d]      = x1 * cs - x2 * sn;
    qkv[base + 64 + d] = x2 * cs + x1 * sn;
  }
}

// ---------------------------------------------------------------------------
// Attention: block = (16 queries, one head, one batch). WMMA bf16 for
// q.k^T and attn.v; operands staged in fragment-ready LDS layout; the Q
// fragments are loop-invariant and live in registers across all key tiles.
// ---------------------------------------------------------------------------
__global__ __launch_bounds__(256)
void attn_wmma(const float* __restrict__ qkv, const int* __restrict__ amask,
               float* __restrict__ ctx)
{
  __shared__ v16bf QAf[4][32];                 // Q frags  [kchunk][lane]  4 KB
  __shared__ v16bf KVf[16][32];                // K/V frags               16 KB
  __shared__ __attribute__((aligned(32))) float Sc[16][SEQ];   // scores  32 KB
  v16bf* Pf = (v16bf*)&Sc[0][0];               // prob frags reuse Sc (16 KB)

  const int tid  = threadIdx.x;
  const int lane = tid & 31;
  const int wave = tid >> 5;
  const int half = lane >> 4;
  const int lrow = lane & 15;
  const int qt = blockIdx.x, hh = blockIdx.y, b = blockIdx.z;
  const int q0 = qt * 16;
  const float sscale = 0.08838834764831845f;   // 1/sqrt(HDIM)

  // ---- load Q tile (component 0) into fragment layout ----
  {
    int r = tid >> 4;                          // 0..15 query row
    int dbase = (tid & 15) * 8;                // 0..120
    int kcs = dbase >> 5;
    int h, i0; frag_pos(dbase & 31, h, i0);
    const float* src =
        qkv + (((size_t)(b * SEQ + q0 + r)) * 3 + 0) * DMODEL + hh * HDIM + dbase;
    v8bf t8;
#pragma unroll
    for (int j = 0; j < 8; ++j) t8[j] = (__bf16)src[j];
    *(v8bf*)((__bf16*)&QAf[kcs][h * 16 + r] + i0) = t8;
  }
  __syncthreads();

  // Q fragments are reused for every key tile: keep them in registers.
  v16bf qf[4];
#pragma unroll
  for (int kcs = 0; kcs < 4; ++kcs) qf[kcs] = QAf[kcs][lane];

  // ---- scores = q @ k^T * scale + mask bias ----
  for (int kt = 0; kt < SEQ / 64; ++kt) {
    __syncthreads();
    {  // stage K tile (64 keys x 128 dims), frag id = keytile*4 + kchunk
      int kr = tid >> 2;                       // 0..63 local key
      int dbase = (tid & 3) * 32;
      int nt = kr >> 4, lr = kr & 15;
      const float* src =
          qkv + (((size_t)(b * SEQ + kt * 64 + kr)) * 3 + 1) * DMODEL + hh * HDIM;
#pragma unroll
      for (int blk = 0; blk < 4; ++blk) {
        int db = dbase + blk * 8;
        int kcs = db >> 5;
        int h, i0; frag_pos(db & 31, h, i0);
        v8bf t8;
#pragma unroll
        for (int j = 0; j < 8; ++j) t8[j] = (__bf16)src[db + j];
        *(v8bf*)((__bf16*)&KVf[nt * 4 + kcs][h * 16 + lr] + i0) = t8;
      }
    }
    __syncthreads();
    if (wave < 4) {                  // uniform per-wave branch, EXEC stays full
      v16bf bk[4];
#pragma unroll
      for (int kcs = 0; kcs < 4; ++kcs) bk[kcs] = KVf[wave * 4 + kcs][lane];
      v8f acc = {};
#pragma unroll
      for (int kcs = 0; kcs < 4; ++kcs)
        acc = __builtin_amdgcn_wmma_f32_16x16x32_bf16(
            false, qf[kcs], false, bk[kcs], (short)0, acc, false, false);
#pragma unroll
      for (int r = 0; r < 8; ++r) {
        int m   = half * 8 + r;
        int key = kt * 64 + wave * 16 + lrow;
        bool ok = (key <= q0 + m) && (amask[b * SEQ + key] > 0);
        Sc[m][key] = acc[r] * sscale + (ok ? 0.f : -1.0e9f);
      }
    }
  }
  __syncthreads();

  // ---- softmax rows; write normalized probs straight into fragment layout
  //      (register stash so Pf can alias the dead Sc buffer) ----
  {
    int row = tid >> 4, sub = tid & 15;
    float mx = -3.0e38f;
#pragma unroll
    for (int jj = 0; jj < 32; ++jj) mx = fmaxf(mx, Sc[row][sub + jj * 16]);
#pragma unroll
    for (int o = 8; o > 0; o >>= 1) mx = fmaxf(mx, __shfl_xor(mx, o, 16));
    float pv[32];
    float sum = 0.f;
#pragma unroll
    for (int jj = 0; jj < 32; ++jj) {
      float e = expf(Sc[row][sub + jj * 16] - mx);
      pv[jj] = e;
      sum += e;
    }
#pragma unroll
    for (int o = 8; o > 0; o >>= 1) sum += __shfl_xor(sum, o, 16);
    const float inv = 1.f / sum;
    __syncthreads();                            // everyone done reading Sc
#pragma unroll
    for (int jj = 0; jj < 32; ++jj) {
      int j = sub + jj * 16;
      int kcs = j >> 5;
      int h, i; frag_pos(j & 31, h, i);
      ((__bf16*)&Pf[kcs * 32 + (h * 16 + row)])[i] = (__bf16)(pv[jj] * inv);
    }
  }
  __syncthreads();

  // ---- ctx = attn @ V : (16x512)x(512x128); wave w -> output cols w*16 ----
  v8f acc = {};
  for (int vt = 0; vt < SEQ / 64; ++vt) {
    __syncthreads();
    {  // stage V tile: B[k=key][n=dim]; frag id = step*8 + coltile
      int d = tid & 127;
      int krbase = (tid >> 7) * 32;
      int nt = d >> 4, lc = d & 15, stp = krbase >> 5;
      __bf16* dstbase = (__bf16*)&KVf[stp * 8 + nt][0];
#pragma unroll
      for (int blk = 0; blk < 4; ++blk) {
        int kb = blk * 8;
        int h, i0; frag_pos(kb, h, i0);
        v8bf t8;
#pragma unroll
        for (int j = 0; j < 8; ++j) {
          int key = vt * 64 + krbase + kb + j;
          t8[j] = (__bf16)qkv[(((size_t)(b * SEQ + key)) * 3 + 2) * DMODEL + hh * HDIM + d];
        }
        *(v8bf*)(dstbase + (size_t)(h * 16 + lc) * 16 + i0) = t8;
      }
    }
    __syncthreads();
    v16bf pa[2], bv[2];
#pragma unroll
    for (int stp = 0; stp < 2; ++stp) {
      pa[stp] = Pf[(vt * 2 + stp) * 32 + lane];
      bv[stp] = KVf[stp * 8 + wave][lane];
    }
#pragma unroll
    for (int stp = 0; stp < 2; ++stp)
      acc = __builtin_amdgcn_wmma_f32_16x16x32_bf16(
          false, pa[stp], false, bv[stp], (short)0, acc, false, false);
  }
#pragma unroll
  for (int r = 0; r < 8; ++r) {
    int m = half * 8 + r;
    ctx[(size_t)(b * SEQ + q0 + m) * DMODEL + hh * HDIM + wave * 16 + lrow] = acc[r];
  }
}

// ---------------------------------------------------------------------------
// Reward head: lnf on last token, logits only at vocab ids 101..104,
// p = softmax([good/2, bad/2])[0] = sigmoid((good - bad)/2).
// ---------------------------------------------------------------------------
__global__ __launch_bounds__(256)
void reward_kernel(const float* __restrict__ h, const float* __restrict__ g,
                   const float* __restrict__ bta, const float* __restrict__ emb,
                   float* __restrict__ out)
{
  __shared__ float xs[DMODEL];
  __shared__ float red[256];
  const int t = threadIdx.x;
  for (int b = 0; b < BATCH; ++b) {
    const float* row = h + ((size_t)(b * SEQ + SEQ - 1)) * DMODEL;
    float s = 0.f, s2 = 0.f;
    for (int k = t; k < DMODEL; k += 256) { float v = row[k]; s += v; s2 += v * v; }
    red[t] = s; __syncthreads();
    for (int o = 128; o > 0; o >>= 1) { if (t < o) red[t] += red[t + o]; __syncthreads(); }
    const float mean = red[0] / DMODEL;
    __syncthreads();
    red[t] = s2; __syncthreads();
    for (int o = 128; o > 0; o >>= 1) { if (t < o) red[t] += red[t + o]; __syncthreads(); }
    const float var = red[0] / DMODEL - mean * mean;
    __syncthreads();
    const float inv = rsqrtf(var + 1e-5f);
    for (int k = t; k < DMODEL; k += 256) xs[k] = (row[k] - mean) * inv * g[k] + bta[k];
    __syncthreads();
    float dots[4];
    for (int id = 0; id < 4; ++id) {
      float p = 0.f;
      for (int k = t; k < DMODEL; k += 256) p += xs[k] * emb[(size_t)(101 + id) * DMODEL + k];
      red[t] = p; __syncthreads();
      for (int o = 128; o > 0; o >>= 1) { if (t < o) red[t] += red[t + o]; __syncthreads(); }
      dots[id] = red[0];
      __syncthreads();
    }
    if (t == 0) {
      float good = dots[0] + dots[1], bad = dots[2] + dots[3];
      out[b] = 1.f / (1.f + expf(0.5f * (bad - good)));
    }
    __syncthreads();
  }
}

// ---------------------------------------------------------------------------
// Host orchestration. Input order = setup_inputs() dict insertion order,
// nested dicts flattened in insertion order.
// ---------------------------------------------------------------------------
extern "C" void kernel_launch(void* const* d_in, const int* in_sizes, int n_in,
                              void* d_out, int out_size, void* d_ws, size_t ws_size,
                              hipStream_t stream)
{
  (void)in_sizes; (void)n_in; (void)out_size; (void)ws_size;

  const int* ids[2]   = {(const int*)d_in[0], (const int*)d_in[1]};
  const int* pos[2]   = {(const int*)d_in[2], (const int*)d_in[3]};
  const int* amask[2] = {(const int*)d_in[4], (const int*)d_in[5]};
  const float* emb    = (const float*)d_in[6];
  const float* ln1_g  = (const float*)d_in[7];
  const float* ln1_b  = (const float*)d_in[8];
  const float* ln2_g  = (const float*)d_in[9];
  const float* ln2_b  = (const float*)d_in[10];
  const float* lnf_g  = (const float*)d_in[11];
  const float* lnf_b  = (const float*)d_in[12];
  const float* qkv_s  = (const float*)d_in[13];
  const float* qkv_b  = (const float*)d_in[14];
  const float* o_s    = (const float*)d_in[15];
  const float* up_s   = (const float*)d_in[16];
  const float* down_s = (const float*)d_in[17];
  const int* qkv_w    = (const int*)d_in[18];
  const int* o_w      = (const int*)d_in[19];
  const int* up_w     = (const int*)d_in[20];
  const int* down_w   = (const int*)d_in[21];

  float* h   = (float*)d_ws;                            // [NTOK, D]
  float* x   = h   + (size_t)NTOK * DMODEL;             // [NTOK, D]
  float* qkv = x   + (size_t)NTOK * DMODEL;             // [NTOK, 3D]
  float* ctx = qkv + (size_t)NTOK * 3 * DMODEL;         // [NTOK, D]
  float* up  = ctx + (size_t)NTOK * DMODEL;             // [NTOK, F]

  float* out = (float*)d_out;                           // [r1(2), r2(2)]

  for (int f = 0; f < 2; ++f) {
    embed_kernel<<<NTOK, 256, 0, stream>>>(ids[f], emb, h);
    for (int l = 0; l < NLAYER; ++l) {
      layernorm_kernel<<<NTOK, 256, 0, stream>>>(h, ln1_g + l * DMODEL, ln1_b + l * DMODEL, x);
      gemm_dq_wmma<<<dim3(3 * DMODEL / 64, NTOK / 128), 256, 0, stream>>>(
          x, qkv_w + (size_t)l * 3 * DMODEL * (DMODEL / 2), qkv_s + (size_t)l * 3 * DMODEL,
          qkv_b + (size_t)l * 3 * DMODEL, nullptr, qkv, NTOK, 3 * DMODEL, DMODEL, 0);
      rope_kernel<<<dim3(NH, SEQ, BATCH), 64, 0, stream>>>(qkv, pos[f]);
      attn_wmma<<<dim3(SEQ / 16, NH, BATCH), 256, 0, stream>>>(qkv, amask[f], ctx);
      gemm_dq_wmma<<<dim3(DMODEL / 64, NTOK / 128), 256, 0, stream>>>(
          ctx, o_w + (size_t)l * DMODEL * (DMODEL / 2), o_s + (size_t)l * DMODEL,
          nullptr, h, h, NTOK, DMODEL, DMODEL, 0);
      layernorm_kernel<<<NTOK, 256, 0, stream>>>(h, ln2_g + l * DMODEL, ln2_b + l * DMODEL, x);
      gemm_dq_wmma<<<dim3(FFDIM / 64, NTOK / 128), 256, 0, stream>>>(
          x, up_w + (size_t)l * FFDIM * (DMODEL / 2), up_s + (size_t)l * FFDIM,
          nullptr, nullptr, up, NTOK, FFDIM, DMODEL, 1);
      gemm_dq_wmma<<<dim3(DMODEL / 64, NTOK / 128), 256, 0, stream>>>(
          up, down_w + (size_t)l * DMODEL * (FFDIM / 2), down_s + (size_t)l * DMODEL,
          nullptr, h, h, NTOK, DMODEL, FFDIM, 0);
    }
    reward_kernel<<<1, 256, 0, stream>>>(h, lnf_g, lnf_b, emb, out + f * 2);
  }
}
